// MCI_GRU_Cell_47949014892891
// MI455X (gfx1250) — matrix operand order repaired and checked by reading, not compiled, not run on hardware
//
#include <hip/hip_runtime.h>
#include <hip/hip_bf16.h>

typedef __attribute__((ext_vector_type(16))) __bf16 v16bf;
typedef __attribute__((ext_vector_type(8)))  float  v8f;

#define B_   256
#define T_   512
#define D_   128
#define U_   256
#define NT   64      /* 4 gates * 16 n-tiles */
#define KTU  8       /* 256/32 */
#define KTW  4       /* 128/32 */
#define FRAG_USH 512 /* 32 lanes * 16 bf16 per B-fragment */

/* ---------- helpers ---------- */
__device__ __forceinline__ unsigned short f2bf(float f) {
  unsigned u = __float_as_uint(f);
  unsigned r = (u + 0x7FFFu + ((u >> 16) & 1u)) >> 16;
  return (unsigned short)r;
}
__device__ __forceinline__ float bf2f(unsigned short h) {
  return __uint_as_float(((unsigned)h) << 16);
}
__device__ __forceinline__ float tanh_fast(float x) {
#if __has_builtin(__builtin_amdgcn_tanhf)
  return __builtin_amdgcn_tanhf(x);
#else
  return tanhf(x);
#endif
}
__device__ __forceinline__ float sigm_fast(float x) {
  return 0.5f + 0.5f * tanh_fast(0.5f * x);
}
__device__ __forceinline__ v8f wmma_bf16(v16bf a, v16bf b, v8f c) {
  return __builtin_amdgcn_wmma_f32_16x16x32_bf16(false, a, false, b, (short)0, c,
                                                 false, false);
}
/* CDNA5 split work-group barrier: signal (after LDS stores drain) ... wait.
   Syntax matches compiler-emitted `s_barrier_signal -1` / `s_barrier_wait -1`. */
__device__ __forceinline__ void wg_signal_after_ds() {
  asm volatile("s_wait_dscnt 0x0\n\ts_barrier_signal -1" ::: "memory");
}
__device__ __forceinline__ void wg_wait() {
  asm volatile("s_barrier_wait -1" ::: "memory");
}

/* ---------- Phase 0: repack fp32 weights into WMMA B-fragment bf16 ---------- */
/* B-frag layout per (ntile, kt): [lane 0..31][16 bf16], element e of lane
   (n + 16*hk) holds M[kt*32 + hk*8 + (e&7) + (e>>3)*16][ntile*16 + n].        */
__global__ void build_frags(const float* __restrict__ Um, const float* __restrict__ Uz,
                            const float* __restrict__ Ur, const float* __restrict__ Uh,
                            const float* __restrict__ Wm, const float* __restrict__ Wz,
                            const float* __restrict__ Wr, const float* __restrict__ Wh,
                            unsigned short* __restrict__ wsU,
                            unsigned short* __restrict__ wsW) {
  int idx = blockIdx.x * blockDim.x + threadIdx.x;
  const int NU = NT * KTU * FRAG_USH;  /* 262144 */
  const int NW = NT * KTW * FRAG_USH;  /* 131072 */
  if (idx < NU) {
    int e = idx & 15, lane = (idx >> 4) & 31, kt = (idx >> 9) & 7, ntile = idx >> 12;
    int g = ntile >> 4;
    int n = ((ntile & 15) << 4) + (lane & 15);
    int k = kt * 32 + ((lane >> 4) << 3) + (e & 7) + ((e >> 3) << 4);
    const float* Ug = (g == 0) ? Um : (g == 1) ? Uz : (g == 2) ? Ur : Uh;
    wsU[idx] = f2bf(Ug[k * U_ + n]);
  } else if (idx < NU + NW) {
    int j = idx - NU;
    int e = j & 15, lane = (j >> 4) & 31, kt = (j >> 9) & 3, ntile = j >> 11;
    int g = ntile >> 4;
    int n = ((ntile & 15) << 4) + (lane & 15);
    int k = kt * 32 + ((lane >> 4) << 3) + (e & 7) + ((e >> 3) << 4);
    const float* Wg = (g == 0) ? Wm : (g == 1) ? Wz : (g == 2) ? Wr : Wh;
    wsW[j] = f2bf(Wg[k * U_ + n]);
  }
}

/* ---------- Phase 1: xp = x @ W_all + b  ->  bf16 C-fragments in ws ---------- */
__global__ __launch_bounds__(512, 1) void input_proj(
    const float* __restrict__ x, const float* __restrict__ bm,
    const float* __restrict__ bz, const float* __restrict__ br,
    const float* __restrict__ bh, const unsigned short* __restrict__ wsW,
    unsigned short* __restrict__ wsXP) {
  __shared__ __align__(16) unsigned short xs[16 * 136]; /* x tile bf16, padded */
  const int bt = blockIdx.x, tc = blockIdx.y;
  const int tid = threadIdx.x, w = tid >> 5, lane = tid & 31;
  const int c = lane & 15, hf = lane >> 4;

  /* W B-fragments for this wave's n-columns, all 4 gates, register-resident */
  v16bf wf[4][KTW];
#pragma unroll
  for (int g = 0; g < 4; ++g)
#pragma unroll
    for (int kt = 0; kt < KTW; ++kt)
      wf[g][kt] = *(const v16bf*)(wsW + (((g * 16 + w) * KTW + kt) * FRAG_USH) + lane * 16);

  float bias[4];
  bias[0] = bm[w * 16 + c]; bias[1] = bz[w * 16 + c];
  bias[2] = br[w * 16 + c]; bias[3] = bh[w * 16 + c];

  for (int i = 0; i < 8; ++i) {
    const int tt = tc * 8 + i;
    /* stage x tile [16 rows][128] as bf16 */
#pragma unroll
    for (int j = 0; j < 4; ++j) {
      int li = tid + j * 512;
      int row = li >> 7, k = li & 127;
      xs[row * 136 + k] = f2bf(x[(size_t)(bt * 16 + row) * (T_ * D_) + tt * D_ + k]);
    }
    __syncthreads();

    v8f acc[4];
#pragma unroll
    for (int g = 0; g < 4; ++g) acc[g] = (v8f){0.f, 0.f, 0.f, 0.f, 0.f, 0.f, 0.f, 0.f};

#pragma unroll
    for (int kt = 0; kt < KTW; ++kt) {
      int k0 = kt * 32 + hf * 8;
      union { uint4 q[2]; v16bf v; } af;
      af.q[0] = *(const uint4*)(&xs[c * 136 + k0]);
      af.q[1] = *(const uint4*)(&xs[c * 136 + k0 + 16]);
#pragma unroll
      for (int g = 0; g < 4; ++g) acc[g] = wmma_bf16(af.v, wf[g][kt], acc[g]);
    }
    __syncthreads();

#pragma unroll
    for (int g = 0; g < 4; ++g) {
      union { unsigned short s[8]; uint4 q; } pk;
#pragma unroll
      for (int v = 0; v < 8; ++v) pk.s[v] = f2bf(acc[g][v] + bias[g]);
      size_t idx = ((size_t)(bt * T_ + tt) * NT + (g * 16 + w)) * 256 + lane * 8;
      *(uint4*)(wsXP + idx) = pk.q;
    }
  }
}

/* ---------- Phase 2: recurrent scan (persistent per batch tile) ---------- */
__global__ __launch_bounds__(512, 1) void gru_scan(
    const unsigned short* __restrict__ wsU,
    const unsigned short* __restrict__ wsXP, float* __restrict__ out) {
  extern __shared__ __align__(32) char smem[];
  unsigned short* uLds = (unsigned short*)smem;              /* gates m,z: 262144 B */
  unsigned short* hF   = (unsigned short*)(smem + 262144);   /* h  A-frags: 8192 B  */
  unsigned short* rhF  = (unsigned short*)(smem + 262144 + 8192); /* r*h A-frags */

  const int bt = blockIdx.x, tid = threadIdx.x, w = tid >> 5, lane = tid & 31;
  const int c = lane & 15, hf = lane >> 4;

  /* cooperative copy: U fragments for gates 0,1 (m,z) into LDS */
  for (int i = tid; i < 262144 / 16; i += 512)
    ((uint4*)uLds)[i] = ((const uint4*)wsU)[i];
  /* zero h / rh fragment buffers */
  for (int i = tid; i < 16384 / 4; i += 512) ((unsigned*)(smem + 262144))[i] = 0u;

  /* register-resident U fragments for gates r (2) and h (3) */
  v16bf ur[KTU], uh[KTU];
#pragma unroll
  for (int kt = 0; kt < KTU; ++kt) {
    ur[kt] = *(const v16bf*)(wsU + (((32 + w) * KTU + kt) * FRAG_USH) + lane * 16);
    uh[kt] = *(const v16bf*)(wsU + (((48 + w) * KTU + kt) * FRAG_USH) + lane * 16);
  }

  float hreg[8];
#pragma unroll
  for (int v = 0; v < 8; ++v) hreg[v] = 0.f;

  /* scatter address for writing this lane's h / r*h values (C-layout ->
     A-fragment layout).  k = 16*w + c  within k-tile kt0 = w/2.            */
  const int klocal = ((w & 1) << 4) + c;
  const int kt0 = w >> 1;
  const int hf2 = (klocal >> 3) & 1;
  const int e2 = (klocal & 7) + ((klocal >> 4) << 3);
  const int sIdx = (kt0 * 32 + hf2 * 16) * 16 + e2; /* ushort index; + row*16 */

  __syncthreads();

  const unsigned short* xpb = wsXP + (size_t)(bt * T_) * NT * 256;
  uint4 xq[4], xqn[4];
#pragma unroll
  for (int g = 0; g < 4; ++g)
    xq[g] = *(const uint4*)(xpb + ((size_t)(0 * NT + g * 16 + w)) * 256 + lane * 8);

  for (int t = 0; t < T_; ++t) {
    const int tn = (t + 1 < T_) ? (t + 1) : t;
    const int tp = (t + 8 < T_) ? (t + 8) : (T_ - 1);
#pragma unroll
    for (int g = 0; g < 4; ++g) { /* prefetch next step's input projections */
      xqn[g] = *(const uint4*)(xpb + ((size_t)(tn * NT + g * 16 + w)) * 256 + lane * 8);
      /* deep prefetch (global_prefetch_b8): stage t+8 into L2 ahead of use */
      __builtin_prefetch(xpb + ((size_t)(tp * NT + g * 16 + w)) * 256 + lane * 8, 0, 1);
    }

    v8f accM = (v8f){0.f,0.f,0.f,0.f,0.f,0.f,0.f,0.f};
    v8f accZ = accM, accR = accM;
#pragma unroll
    for (int kt = 0; kt < KTU; ++kt) {
      v16bf a  = *(const v16bf*)(hF + kt * FRAG_USH + lane * 16);
      v16bf bm_ = *(const v16bf*)(uLds + (((0 * 16 + w) * KTU + kt) * FRAG_USH) + lane * 16);
      v16bf bz_ = *(const v16bf*)(uLds + (((1 * 16 + w) * KTU + kt) * FRAG_USH) + lane * 16);
      accM = wmma_bf16(a, bm_, accM);
      accZ = wmma_bf16(a, bz_, accZ);
      accR = wmma_bf16(a, ur[kt], accR);
    }

    float zv[8], zhm[8];
#pragma unroll
    for (int v = 0; v < 8; ++v) {
      const unsigned* q0 = (const unsigned*)&xq[0];
      const unsigned* q1 = (const unsigned*)&xq[1];
      const unsigned* q2 = (const unsigned*)&xq[2];
      float xm = bf2f((unsigned short)((q0[v >> 1] >> ((v & 1) * 16)) & 0xFFFFu));
      float xz = bf2f((unsigned short)((q1[v >> 1] >> ((v & 1) * 16)) & 0xFFFFu));
      float xr = bf2f((unsigned short)((q2[v >> 1] >> ((v & 1) * 16)) & 0xFFFFu));
      float mv = tanh_fast(accM[v] + xm);
      zv[v] = sigm_fast(accZ[v] + xz);
      float rv = sigm_fast(accR[v] + xr);
      zhm[v] = zv[v] * hreg[v] + mv; /* pre-combine z*h + m off the critical path */
      rhF[sIdx + (8 * hf + v) * 16] = f2bf(rv * hreg[v]);
    }
    wg_signal_after_ds(); /* all r*h fragments drained to LDS */

    /* barrier shadow: unpack x_h projections (independent of other waves) */
    float xh[8];
#pragma unroll
    for (int v = 0; v < 8; ++v) {
      const unsigned* q3 = (const unsigned*)&xq[3];
      xh[v] = bf2f((unsigned short)((q3[v >> 1] >> ((v & 1) * 16)) & 0xFFFFu));
    }
    wg_wait();

    v8f accH = (v8f){0.f,0.f,0.f,0.f,0.f,0.f,0.f,0.f};
#pragma unroll
    for (int kt = 0; kt < KTU; ++kt) {
      v16bf a2 = *(const v16bf*)(rhF + kt * FRAG_USH + lane * 16);
      accH = wmma_bf16(a2, uh[kt], accH);
    }

    float hn[8];
#pragma unroll
    for (int v = 0; v < 8; ++v) {
      float hc = tanh_fast(accH[v] + xh[v]);
      hn[v] = zhm[v] + (1.f - zv[v]) * hc;
      hreg[v] = hn[v];
      hF[sIdx + (8 * hf + v) * 16] = f2bf(hn[v]);
    }
    wg_signal_after_ds(); /* h(t+1) fragments drained to LDS */

    /* barrier shadow: global output stores + double-buffer swap */
#pragma unroll
    for (int v = 0; v < 8; ++v) {
      int row = 8 * hf + v;
      out[((size_t)(bt * 16 + row) * T_ + t) * U_ + w * 16 + c] = hn[v];
    }
#pragma unroll
    for (int g = 0; g < 4; ++g) xq[g] = xqn[g];
    wg_wait();
  }
}

/* ---------- launch ---------- */
extern "C" void kernel_launch(void* const* d_in, const int* in_sizes, int n_in,
                              void* d_out, int out_size, void* d_ws, size_t ws_size,
                              hipStream_t stream) {
  const float* x  = (const float*)d_in[0];
  const float* Wm = (const float*)d_in[1];
  const float* Um = (const float*)d_in[2];
  const float* bm = (const float*)d_in[3];
  const float* Wz = (const float*)d_in[4];
  const float* Uz = (const float*)d_in[5];
  const float* bz = (const float*)d_in[6];
  const float* Wr = (const float*)d_in[7];
  const float* Ur = (const float*)d_in[8];
  const float* br = (const float*)d_in[9];
  const float* Wh = (const float*)d_in[10];
  const float* Uh = (const float*)d_in[11];
  const float* bh = (const float*)d_in[12];

  unsigned short* wsU  = (unsigned short*)d_ws;           /* 262144 ush */
  unsigned short* wsW  = wsU + NT * KTU * FRAG_USH;       /* 131072 ush */
  unsigned short* wsXP = wsW + NT * KTW * FRAG_USH;       /* 134217728 ush */

  /* Phase 0: weight fragments */
  build_frags<<<1536, 256, 0, stream>>>(Um, Uz, Ur, Uh, Wm, Wz, Wr, Wh, wsU, wsW);

  /* Phase 1: input projections (16 batch tiles x 64 time chunks of 8) */
  input_proj<<<dim3(16, 64), 512, 0, stream>>>(x, bm, bz, br, bh, wsW, wsXP);

  /* Phase 2: recurrent scan, one WG per 16-row batch tile; 256KB+16KB LDS */
  gru_scan<<<16, 512, 262144 + 16384, stream>>>(wsU, wsXP, (float*)d_out);
}
